// RBFMin_36773509989120
// MI455X (gfx1250) — compile-verified, matrix operand order checked by reading, not run on hardware
//
#include <hip/hip_runtime.h>

typedef __attribute__((ext_vector_type(16))) __bf16 v16bf;
typedef __attribute__((ext_vector_type(8)))  float  v8f;

#define B_DIM 16
#define S_DIM 4096
#define F_DIM 64
#define O_DIM 256
#define NT 2                        // 16-col N tiles per wave (32 stars)
#define WAVES 8                     // 8 waves x 32 cols = full O per workgroup
#define S_SPLITS 32
#define ROWS_PER_WG 128             // WAVES * 16, one produced slab per wave
#define NSLAB 8

// LDS layout: A fragments with 48B lane stride -> ds_load_b128 hits all 64 banks
#define LANE_STRIDE 48
#define FRAG_BYTES (32 * LANE_STRIDE)    // 1536 per fragment (32 lanes)
#define SLAB_BYTES (4 * FRAG_BYTES)      // hi0, lo0, hi1, lo1
#define X2_BASE (NSLAB * SLAB_BYTES)     // 49152
#define LDS_BYTES (X2_BASE + NSLAB * 64) // + 16 floats of x2 per slab

// Load 16 fp32 values of one row in WMMA-A/B K-order for one 32-wide K chunk:
// elements [0..7] = row[f0 .. f0+7], [8..15] = row[f0+16 .. f0+23].
// Splits each value into bf16 hi + bf16 lo (x ~= hi + lo) and accumulates x^2.
__device__ inline void load_frag(const float* __restrict__ rowp, int f0,
                                 v16bf& hi, v16bf& lo, float& ss) {
  float4 a0 = *(const float4*)(rowp + f0);
  float4 a1 = *(const float4*)(rowp + f0 + 4);
  float4 b0 = *(const float4*)(rowp + f0 + 16);
  float4 b1 = *(const float4*)(rowp + f0 + 20);
  float f[16] = {a0.x, a0.y, a0.z, a0.w, a1.x, a1.y, a1.z, a1.w,
                 b0.x, b0.y, b0.z, b0.w, b1.x, b1.y, b1.z, b1.w};
#pragma unroll
  for (int i = 0; i < 16; i += 2) {   // pairwise -> v_cvt_pk_bf16_f32
    float x0 = f[i], x1 = f[i + 1];
    ss += x0 * x0;
    ss += x1 * x1;
    __bf16 h0 = (__bf16)x0;
    __bf16 h1 = (__bf16)x1;
    hi[i]     = h0;
    hi[i + 1] = h1;
    lo[i]     = (__bf16)(x0 - (float)h0);
    lo[i + 1] = (__bf16)(x1 - (float)h1);
  }
}

__device__ inline void lds_store_frag(void* p, v16bf v) {
  union { v16bf v; uint4 q[2]; } u; u.v = v;
  *(uint4*)p = u.q[0];
  *(uint4*)((char*)p + 16) = u.q[1];
}
__device__ inline v16bf lds_load_frag(const void* p) {
  union { v16bf v; uint4 q[2]; } u;
  u.q[0] = *(const uint4*)p;
  u.q[1] = *(const uint4*)((const char*)p + 16);
  return u.v;
}

__global__ void init_out_kernel(unsigned int* __restrict__ out, int n) {
  int i = blockIdx.x * blockDim.x + threadIdx.x;
  if (i < n) out[i] = 0x7F800000u;  // +inf bit pattern
}

__global__ __launch_bounds__(256, 1)
void rbfmin_kernel(const float* __restrict__ inputs,
                   const float* __restrict__ stars,
                   unsigned int* __restrict__ out) {
  __shared__ __attribute__((aligned(16))) unsigned char lds[LDS_BYTES];

  const int tid  = threadIdx.x;
  const int wave = tid >> 5;
  const int lane = tid & 31;
  const int col  = lane & 15;   // M (A) / N (B) index within tile
  const int h    = lane >> 4;   // K-half selector
  const int b     = blockIdx.y;
  const int obase = wave * (NT * 16);          // this wave's 32 star columns
  const int sbase = blockIdx.x * ROWS_PER_WG;  // this WG's 128 point rows

  // ---- Preload B fragments (stars) in registers + per-column ||c||^2 ----
  v16bf bh[NT][2], bl[NT][2];
  float c2[NT];
#pragma unroll
  for (int t = 0; t < NT; ++t) {
    const float* srow = stars + (size_t)(obase + t * 16 + col) * F_DIM;
    float ss = 0.0f;
    load_frag(srow,      8 * h, bh[t][0], bl[t][0], ss);
    load_frag(srow, 32 + 8 * h, bh[t][1], bl[t][1], ss);
    c2[t] = ss + __shfl_xor(ss, 16, 32);   // full ||star||^2 for column col
  }

  // ---- Produce phase: wave w converts slab w (16 rows) into LDS ----
  {
    const int s = sbase + wave * 16 + col;
    const float* arow = inputs + ((size_t)b * S_DIM + s) * F_DIM;
    v16bf ah0, al0, ah1, al1;
    float ss = 0.0f;
    load_frag(arow,      8 * h, ah0, al0, ss);
    load_frag(arow, 32 + 8 * h, ah1, al1, ss);
    const float x2 = ss + __shfl_xor(ss, 16, 32);  // ||x_row||^2, row = col

    unsigned char* pb = lds + wave * SLAB_BYTES + lane * LANE_STRIDE;
    lds_store_frag(pb + 0 * FRAG_BYTES, ah0);
    lds_store_frag(pb + 1 * FRAG_BYTES, al0);
    lds_store_frag(pb + 2 * FRAG_BYTES, ah1);
    lds_store_frag(pb + 3 * FRAG_BYTES, al1);
    if (lane < 16)
      *(float*)(lds + X2_BASE + wave * 64 + lane * 4) = x2;
  }
  __syncthreads();

  // running min of (||x||^2 - 2 x.c); ||c||^2 is column-constant -> folded at end
  float runmin[NT][8];
#pragma unroll
  for (int t = 0; t < NT; ++t)
#pragma unroll
    for (int j = 0; j < 8; ++j) runmin[t][j] = 3.4e38f;

  // ---- Consume phase: every wave walks all 8 slabs from LDS ----
  for (int sl = 0; sl < NSLAB; ++sl) {
    const unsigned char* qb = lds + sl * SLAB_BYTES + lane * LANE_STRIDE;
    v16bf ah0 = lds_load_frag(qb + 0 * FRAG_BYTES);
    v16bf al0 = lds_load_frag(qb + 1 * FRAG_BYTES);
    v16bf ah1 = lds_load_frag(qb + 2 * FRAG_BYTES);
    v16bf al1 = lds_load_frag(qb + 3 * FRAG_BYTES);

    const float4 xa = *(const float4*)(lds + X2_BASE + sl * 64 + h * 32);
    const float4 xb = *(const float4*)(lds + X2_BASE + sl * 64 + h * 32 + 16);
    const float x2v[8] = {xa.x, xa.y, xa.z, xa.w, xb.x, xb.y, xb.z, xb.w};

#pragma unroll
    for (int t = 0; t < NT; ++t) {
      v8f acc = {0.f, 0.f, 0.f, 0.f, 0.f, 0.f, 0.f, 0.f};
      // bf16x3 split-precision dot product: hi*hi + hi*lo + lo*hi
      acc = __builtin_amdgcn_wmma_f32_16x16x32_bf16(false, ah0, false, bh[t][0], (short)0, acc, false, false);
      acc = __builtin_amdgcn_wmma_f32_16x16x32_bf16(false, ah0, false, bl[t][0], (short)0, acc, false, false);
      acc = __builtin_amdgcn_wmma_f32_16x16x32_bf16(false, al0, false, bh[t][0], (short)0, acc, false, false);
      acc = __builtin_amdgcn_wmma_f32_16x16x32_bf16(false, ah1, false, bh[t][1], (short)0, acc, false, false);
      acc = __builtin_amdgcn_wmma_f32_16x16x32_bf16(false, ah1, false, bl[t][1], (short)0, acc, false, false);
      acc = __builtin_amdgcn_wmma_f32_16x16x32_bf16(false, al1, false, bh[t][1], (short)0, acc, false, false);
#pragma unroll
      for (int j = 0; j < 8; ++j) {
        // d2 - c2 = x2 - 2*dot : single v_fma + v_min per element
        float dpart = __builtin_fmaf(acc[j], -2.0f, x2v[j]);
        runmin[t][j] = fminf(runmin[t][j], dpart);
      }
    }
  }

  // ---- Reduce: 8 M-rows in-lane, row-halves across lane^16, then atomic ----
#pragma unroll
  for (int t = 0; t < NT; ++t) {
    float m = runmin[t][0];
#pragma unroll
    for (int j = 1; j < 8; ++j) m = fminf(m, runmin[t][j]);
    m = fminf(m, __shfl_xor(m, 16, 32));
    float dist = sqrtf(fmaxf(m + c2[t], 0.0f));    // fold ||c||^2 back in
    if (lane < 16) {
      // dist >= 0 so uint bit-pattern order == float order
      atomicMin(out + ((size_t)b * O_DIM + obase + t * 16 + lane),
                __float_as_uint(dist));
    }
  }
}

extern "C" void kernel_launch(void* const* d_in, const int* in_sizes, int n_in,
                              void* d_out, int out_size, void* d_ws, size_t ws_size,
                              hipStream_t stream) {
  (void)in_sizes; (void)n_in; (void)d_ws; (void)ws_size; (void)out_size;
  const float* inputs = (const float*)d_in[0];
  const float* stars  = (const float*)d_in[1];
  unsigned int* out   = (unsigned int*)d_out;

  init_out_kernel<<<(B_DIM * O_DIM + 255) / 256, 256, 0, stream>>>(out, B_DIM * O_DIM);

  dim3 grid(S_SPLITS, B_DIM);
  rbfmin_kernel<<<grid, dim3(256), 0, stream>>>(inputs, stars, out);
}
